// MutualEncoder_962072674785
// MI455X (gfx1250) — matrix operand order repaired and checked by reading, not compiled, not run on hardware
//
#include <hip/hip_runtime.h>

typedef __attribute__((ext_vector_type(16))) _Float16 v16h;
typedef __attribute__((ext_vector_type(8)))  _Float16 v8h;
typedef __attribute__((ext_vector_type(8)))  float    v8f;

#define COL_DIM 4096
#define ROW_DIM 1024
#define KNN_E   16384
#define GENET_E 131072

#define BM  128
#define BNN 128
#define BK  32
#define LDH 40   // padded LDS row stride in halves (80B, 16B aligned, conflict-spread)

// ---------------- utility kernels ----------------

__global__ void fill_k(float* __restrict__ p, float v, size_t n) {
  size_t i = (size_t)blockIdx.x * blockDim.x + threadIdx.x;
  size_t stride = (size_t)gridDim.x * blockDim.x;
  for (; i < n; i += stride) p[i] = v;
}

// in[R][C] -> out[C][R]
__global__ __launch_bounds__(256) void transpose_k(const float* __restrict__ in,
                                                   float* __restrict__ out,
                                                   int R, int C) {
  __shared__ float t[32][33];
  const int c0 = blockIdx.x * 32, r0 = blockIdx.y * 32;
  const int tx = threadIdx.x, ty = threadIdx.y;
#pragma unroll
  for (int i = 0; i < 4; ++i)
    t[ty + i * 8][tx] = in[(size_t)(r0 + ty + i * 8) * C + (c0 + tx)];
  __syncthreads();
#pragma unroll
  for (int i = 0; i < 4; ++i)
    out[(size_t)(c0 + ty + i * 8) * R + (r0 + tx)] = t[tx][ty + i * 8];
}

__global__ void count_k(const int* __restrict__ dst, float* __restrict__ cnt, int E) {
  int i = blockIdx.x * blockDim.x + threadIdx.x;
  if (i < E) atomicAdd(&cnt[dst[i]], 1.0f);
}

// one block per edge: gather src row (float4) and atomically add into dst row
__global__ __launch_bounds__(256) void scatter_k(const float* __restrict__ feat,
                                                 const int* __restrict__ src,
                                                 const int* __restrict__ dst,
                                                 float* __restrict__ agg, int d) {
  const int e = blockIdx.x;
  const int s = src[e], t = dst[e];
  const float4* fs = (const float4*)(feat + (size_t)s * d);
  float* p = agg + (size_t)t * d;
  for (int c = threadIdx.x; c < (d >> 2); c += blockDim.x) {
    float4 v = fs[c];
    atomicAdd(p + 4 * c + 0, v.x);
    atomicAdd(p + 4 * c + 1, v.y);
    atomicAdd(p + 4 * c + 2, v.z);
    atomicAdd(p + 4 * c + 3, v.w);
  }
}

// one block per node row: agg /= max(cnt,1)
__global__ __launch_bounds__(256) void meandiv_k(float* __restrict__ agg,
                                                 const float* __restrict__ cnt, int d) {
  const int row = blockIdx.x;
  const float inv = 1.0f / fmaxf(cnt[row], 1.0f);
  float* p = agg + (size_t)row * d;
  for (int c = threadIdx.x; c < d; c += blockDim.x) p[c] *= inv;
}

// pack 8 f32 -> 8 f16
static __device__ __forceinline__ v8h cvt8(float4 a, float4 b) {
  v8h h;
  h[0] = (_Float16)a.x; h[1] = (_Float16)a.y; h[2] = (_Float16)a.z; h[3] = (_Float16)a.w;
  h[4] = (_Float16)b.x; h[5] = (_Float16)b.y; h[6] = (_Float16)b.z; h[7] = (_Float16)b.w;
  return h;
}

// ---------------- fused SAGE GEMM ----------------
// out[M,N] = leaky( A1[M,K] * W1[N,K]^T + A2[M,K] * W2[N,K]^T + bias[N] )
// Block tile 128x128, 8 waves (4 in M x 2 in N), wave tile 32x64 =
// 2x4 v_wmma_f32_16x16x32_f16 accumulators, K-step 32.
// Double-buffered LDS with f32->f16 register staging; one barrier per step;
// the Wl and Wr passes are flattened into a single 2*(K/32)-step pipeline.
__global__ __launch_bounds__(256)
void sage_gemm_leaky_k(const float* __restrict__ A1, const float* __restrict__ A2,
                       const float* __restrict__ W1, const float* __restrict__ W2,
                       const float* __restrict__ bias, float* __restrict__ out,
                       int M, int N, int K) {
  __shared__ alignas(16) _Float16 As[2][BM * LDH];
  __shared__ alignas(16) _Float16 Bs[2][BNN * LDH];

  const int tid  = threadIdx.x;
  const int lane = tid & 31;
  const int wave = tid >> 5;
  const int wm   = wave & 3;   // wave M position (0..3), 32 rows each
  const int wn   = wave >> 2;  // wave N position (0..1), 64 cols each
  const size_t mBlock = (size_t)blockIdx.y * BM;
  const size_t nBlock = (size_t)blockIdx.x * BNN;

  // staging map: 256 threads, each moves 16 A floats + 16 W floats per step
  const int srow = tid >> 1;        // 0..127
  const int scol = (tid & 1) * 16;  // 0,16

  // fragment read positions (ISA 16-bit A 16x32 layout)
  const int fm  = lane & 15;
  const int akb = (lane >> 4) * 8;    // A frag: runs [akb,akb+8) and [akb+16,akb+24)
  const int bkb = (lane >> 4) * 8;    // B frag: runs [bkb,bkb+8) and [bkb+16,bkb+24)

  v8f acc[2][4] = {};

  const int kSteps = K / BK;
  const int steps  = 2 * kSteps;   // pass 0: (A1,W1); pass 1: (A2,W2)

  v8h ra0, ra1, rb0, rb1;  // staged f16 registers

  auto load_regs = [&](int s) {
    const int pass = (s >= kSteps);
    const int k0 = (s - (pass ? kSteps : 0)) * BK;
    const float* __restrict__ a =
        (pass ? A2 : A1) + (mBlock + srow) * (size_t)K + (k0 + scol);
    const float* __restrict__ b =
        (pass ? W2 : W1) + (nBlock + srow) * (size_t)K + (k0 + scol);
    __builtin_prefetch(a + BK, 0, 3);
    __builtin_prefetch(b + BK, 0, 3);
    float4 a0 = *(const float4*)(a);
    float4 a1 = *(const float4*)(a + 4);
    float4 a2 = *(const float4*)(a + 8);
    float4 a3 = *(const float4*)(a + 12);
    float4 b0 = *(const float4*)(b);
    float4 b1 = *(const float4*)(b + 4);
    float4 b2 = *(const float4*)(b + 8);
    float4 b3 = *(const float4*)(b + 12);
    ra0 = cvt8(a0, a1); ra1 = cvt8(a2, a3);
    rb0 = cvt8(b0, b1); rb1 = cvt8(b2, b3);
  };

  auto store_lds = [&](int p) {
    *(v8h*)&As[p][srow * LDH + scol]     = ra0;
    *(v8h*)&As[p][srow * LDH + scol + 8] = ra1;
    *(v8h*)&Bs[p][srow * LDH + scol]     = rb0;
    *(v8h*)&Bs[p][srow * LDH + scol + 8] = rb1;
  };

  // prologue
  load_regs(0);
  store_lds(0);
  __syncthreads();

  for (int s = 0; s < steps; ++s) {
    const int p = s & 1;
    if (s + 1 < steps) load_regs(s + 1);  // overlap next global loads with compute

    v16h af[2], bf[4];
#pragma unroll
    for (int i = 0; i < 2; ++i) {
      const int r = wm * 32 + i * 16 + fm;
      v8h lo = *(const v8h*)&As[p][r * LDH + akb];
      v8h hi = *(const v8h*)&As[p][r * LDH + akb + 16];
#pragma unroll
      for (int e = 0; e < 8; ++e) { af[i][e] = lo[e]; af[i][8 + e] = hi[e]; }
    }
#pragma unroll
    for (int j = 0; j < 4; ++j) {
      const int r = wn * 64 + j * 16 + fm;
      v8h lo = *(const v8h*)&Bs[p][r * LDH + bkb];
      v8h hi = *(const v8h*)&Bs[p][r * LDH + bkb + 16];
#pragma unroll
      for (int e = 0; e < 8; ++e) { bf[j][e] = lo[e]; bf[j][8 + e] = hi[e]; }
    }
#pragma unroll
    for (int i = 0; i < 2; ++i)
#pragma unroll
      for (int j = 0; j < 4; ++j)
        acc[i][j] = __builtin_amdgcn_wmma_f32_16x16x32_f16(
            false, af[i], false, bf[j], (short)0, acc[i][j], false, false);

    if (s + 1 < steps) store_lds(p ^ 1);  // fill the other buffer
    __syncthreads();
  }

  // epilogue: bias + leaky_relu(0.01); D layout: lane n=lane&15, rows r + 8*(lane>>4)
  const int dh = lane >> 4;
  const int dn = lane & 15;
#pragma unroll
  for (int i = 0; i < 2; ++i) {
#pragma unroll
    for (int j = 0; j < 4; ++j) {
      const size_t gm0 = mBlock + wm * 32 + i * 16 + dh * 8;
      const size_t gn  = nBlock + wn * 64 + j * 16 + dn;
      const float bb = bias[gn];
#pragma unroll
      for (int r = 0; r < 8; ++r) {
        float v = acc[i][j][r] + bb;
        v = v > 0.0f ? v : 0.01f * v;
        out[(gm0 + r) * N + gn] = v;
      }
    }
  }
}

// ---------------- driver ----------------

extern "C" void kernel_launch(void* const* d_in, const int* in_sizes, int n_in,
                              void* d_out, int out_size, void* d_ws, size_t ws_size,
                              hipStream_t stream) {
  (void)in_sizes; (void)n_in; (void)out_size; (void)ws_size;

  const float* x     = (const float*)d_in[0];
  const float* colWl = (const float*)d_in[1];
  const float* colWr = (const float*)d_in[2];
  const float* colB  = (const float*)d_in[3];
  const float* rowWl = (const float*)d_in[4];
  const float* rowWr = (const float*)d_in[5];
  const float* rowB  = (const float*)d_in[6];
  const int*   knn   = (const int*)d_in[7];
  const int*   genet = (const int*)d_in[8];
  float* out = (float*)d_out;
  float* ws  = (float*)d_ws;

  const size_t NE = (size_t)COL_DIM * ROW_DIM;  // 4M elements
  float* f1  = ws;            // e^T        [ROW_DIM, COL_DIM]
  float* f2  = f1 + NE;       // mean agg   (either shape)
  float* f3  = f2 + NE;       // col-branch GEMM out [ROW_DIM, COL_DIM]
  float* w0  = f3 + NE;       // e after col branch [COL_DIM, ROW_DIM]
  float* w1  = w0 + NE;       // layer output ping  [COL_DIM, ROW_DIM]
  float* cnt = w1 + NE;       // degree counts (<= COL_DIM floats)

  const int* knnS = knn;    const int* knnD = knn + KNN_E;
  const int* genS = genet;  const int* genD = genet + GENET_E;

  const float* cur = x;
  for (int L = 0; L < 3; ++L) {
    const size_t cwo = (size_t)L * COL_DIM * COL_DIM;
    const size_t rwo = (size_t)L * ROW_DIM * ROW_DIM;

    // ---- column branch: SAGE over e^T [ROW_DIM nodes x COL_DIM feats] ----
    transpose_k<<<dim3(ROW_DIM / 32, COL_DIM / 32), dim3(32, 8), 0, stream>>>(
        cur, f1, COL_DIM, ROW_DIM);
    fill_k<<<64, 256, 0, stream>>>(cnt, 0.0f, (size_t)ROW_DIM);
    fill_k<<<2048, 256, 0, stream>>>(f2, 0.0f, NE);
    count_k<<<(KNN_E + 255) / 256, 256, 0, stream>>>(knnD, cnt, KNN_E);
    scatter_k<<<KNN_E, 256, 0, stream>>>(f1, knnS, knnD, f2, COL_DIM);
    meandiv_k<<<ROW_DIM, 256, 0, stream>>>(f2, cnt, COL_DIM);
    sage_gemm_leaky_k<<<dim3(COL_DIM / BNN, ROW_DIM / BM), 256, 0, stream>>>(
        f2, f1, colWl + cwo, colWr + cwo, colB + (size_t)L * COL_DIM, f3,
        ROW_DIM, COL_DIM, COL_DIM);
    transpose_k<<<dim3(COL_DIM / 32, ROW_DIM / 32), dim3(32, 8), 0, stream>>>(
        f3, w0, ROW_DIM, COL_DIM);

    // ---- row branch: SAGE over e [COL_DIM nodes x ROW_DIM feats] ----
    fill_k<<<64, 256, 0, stream>>>(cnt, 0.0f, (size_t)COL_DIM);
    fill_k<<<2048, 256, 0, stream>>>(f2, 0.0f, NE);
    count_k<<<(GENET_E + 255) / 256, 256, 0, stream>>>(genD, cnt, GENET_E);
    scatter_k<<<GENET_E, 256, 0, stream>>>(w0, genS, genD, f2, ROW_DIM);
    meandiv_k<<<COL_DIM, 256, 0, stream>>>(f2, cnt, ROW_DIM);
    float* tgt = (L == 2) ? out : w1;
    sage_gemm_leaky_k<<<dim3(ROW_DIM / BNN, COL_DIM / BM), 256, 0, stream>>>(
        f2, w0, rowWl + rwo, rowWr + rwo, rowB + (size_t)L * ROW_DIM, tgt,
        COL_DIM, ROW_DIM, ROW_DIM);
    cur = tgt;
  }
}
